// StandardROIHeads_33913061769628
// MI455X (gfx1250) — compile-verified, matrix operand order checked by reading, not run on hardware
//
#include <hip/hip_runtime.h>

typedef __attribute__((ext_vector_type(16))) _Float16 v16h;
typedef __attribute__((ext_vector_type(8)))  float    v8f;

#define Bb 8
#define Np 256
#define Sp 1024
#define Vp 1024
#define CsC 256
#define CvC 128
#define KIN 384          // Cv + Cs
#define Hh 512
#define NCLS 10
#define NCOL 18          // 11 + 6 + 1
#define NMS_M 2560       // N * NCLS
#define NMS_P 4096
#define SCORE_T 0.05f
#define NMS_T 0.25f

// d_out float offsets
#define OFF_HEAD    0
#define OFF_BOXES   36864
#define OFF_SCORES  159744
#define OFF_CLASSES 180224
#define OFF_KEEP    200704

// ---------------------------------------------------------------------------
// 1) W1 [384,512] f32  ->  W1T [512][384] f16 (row per output col)
// ---------------------------------------------------------------------------
__global__ void prep_w1_kernel(const float* __restrict__ W1, _Float16* __restrict__ W1T) {
  int idx = blockIdx.x * blockDim.x + threadIdx.x;       // 512*384
  if (idx >= Hh * KIN) return;
  int col = idx / KIN, k = idx % KIN;
  W1T[idx] = (_Float16)W1[k * Hh + col];
}

// ---------------------------------------------------------------------------
// 2) ROI grid pool + voted gather -> featsA f16 [B*N, 384]
//    one block per proposal; seed_xyz in LDS; wave-per-gridpoint argmin
// ---------------------------------------------------------------------------
__global__ void gather_pool_kernel(const float* __restrict__ seed_xyz,
                                   const float* __restrict__ seed_feat,
                                   const float* __restrict__ voted,
                                   const int*   __restrict__ prop_inds,
                                   const float* __restrict__ lbdrfu,
                                   const float* __restrict__ origins,
                                   _Float16* __restrict__ featsA) {
  __shared__ float sx[Sp], sy[Sp], sz[Sp];
  __shared__ int sIdx[27];
  int bn = blockIdx.x;                 // 0..2047
  int b = bn >> 8;
  int tid = threadIdx.x;
  for (int s = tid; s < Sp; s += 256) {
    const float* p = seed_xyz + (b * Sp + s) * 3;
    sx[s] = p[0]; sy[s] = p[1]; sz[s] = p[2];
  }
  float bmn[3], bmx[3];
  {
    const float* L = lbdrfu + bn * 6;
    const float* O = origins + bn * 3;
    #pragma unroll
    for (int d = 0; d < 3; ++d) { bmn[d] = O[d] - L[d]; bmx[d] = O[d] + L[3 + d]; }
  }
  __syncthreads();
  int wave = tid >> 5, lane = tid & 31;
  for (int g = wave; g < 27; g += 8) {
    int gx = g / 9, gy = (g / 3) % 3, gz = g % 3;
    float px = bmn[0] + ((gx + 0.5f) / 3.0f) * (bmx[0] - bmn[0]);
    float py = bmn[1] + ((gy + 0.5f) / 3.0f) * (bmx[1] - bmn[1]);
    float pz = bmn[2] + ((gz + 0.5f) / 3.0f) * (bmx[2] - bmn[2]);
    float bd = 3.4e38f; int bi = 0;
    for (int s = lane; s < Sp; s += 32) {
      float dx = px - sx[s], dy = py - sy[s], dz = pz - sz[s];
      float d2 = dx * dx + dy * dy + dz * dz;
      if (d2 < bd) { bd = d2; bi = s; }
    }
    #pragma unroll
    for (int off = 16; off > 0; off >>= 1) {
      float od = __shfl_xor(bd, off, 32);
      int   oi = __shfl_xor(bi, off, 32);
      if (od < bd || (od == bd && oi < bi)) { bd = od; bi = oi; }
    }
    if (lane == 0) sIdx[g] = bi;
  }
  __syncthreads();
  _Float16* fa = featsA + (long)bn * KIN;
  if (tid < CvC) {
    int v = prop_inds[bn];
    fa[tid] = (_Float16)voted[((long)b * CvC + tid) * Vp + v];
  }
  {
    int c = tid;                       // 0..255 == Cs channels
    const float* sf = seed_feat + ((long)b * CsC + c) * Sp;
    float m = -3.4e38f;
    #pragma unroll
    for (int g = 0; g < 27; ++g) m = fmaxf(m, sf[sIdx[g]]);
    fa[CvC + c] = (_Float16)m;
  }
}

// ---------------------------------------------------------------------------
// 3) WMMA GEMM: h = relu(featsA[2048,384] @ W1 + b1) -> hbuf f32 [2048,512]
//    grid (128, 4), block 256 = 8 waves; wave -> one 16x16 tile, 12 WMMAs
// ---------------------------------------------------------------------------
__global__ void gemm1_kernel(const _Float16* __restrict__ A,
                             const _Float16* __restrict__ Bt,
                             const float* __restrict__ bias,
                             float* __restrict__ C) {
  int wave = threadIdx.x >> 5, lane = threadIdx.x & 31;
  int m = lane & 15, half = lane >> 4;
  int row = blockIdx.x * 16 + m;                       // A row for this lane
  int col = blockIdx.y * 128 + wave * 16 + m;          // B/D column for this lane
  const _Float16* ap = A + (long)row * KIN;
  const _Float16* bp = Bt + (long)col * KIN;
  v8f acc = {};
  #pragma unroll
  for (int k0 = 0; k0 < KIN; k0 += 32) {
    v16h a, bf;
    *reinterpret_cast<uint4*>(&a)        = *reinterpret_cast<const uint4*>(ap + k0 + half * 8);
    *(reinterpret_cast<uint4*>(&a) + 1)  = *reinterpret_cast<const uint4*>(ap + k0 + 16 + half * 8);
    *reinterpret_cast<uint4*>(&bf)       = *reinterpret_cast<const uint4*>(bp + k0 + half * 8);
    *(reinterpret_cast<uint4*>(&bf) + 1) = *reinterpret_cast<const uint4*>(bp + k0 + 16 + half * 8);
    acc = __builtin_amdgcn_wmma_f32_16x16x32_f16(false, a, false, bf,
                                                 (short)0, acc, false, false);
  }
  float bv = bias[col];
  #pragma unroll
  for (int r = 0; r < 8; ++r) {
    int orow = blockIdx.x * 16 + r + 8 * half;
    C[(long)orow * Hh + col] = fmaxf(acc[r] + bv, 0.0f);
  }
}

// ---------------------------------------------------------------------------
// 4) Heads: 18 dots of length 512 per row (wave/row), softmax, box decode
// ---------------------------------------------------------------------------
__global__ void heads_kernel(const float* __restrict__ hbuf,
                             const float* __restrict__ Wc, const float* __restrict__ bc,
                             const float* __restrict__ Wb, const float* __restrict__ bb,
                             const float* __restrict__ Wz, const float* __restrict__ bz,
                             const float* __restrict__ lbdrfu,
                             const float* __restrict__ origins,
                             float* __restrict__ out,
                             float* __restrict__ cboxes,
                             float* __restrict__ cscores) {
  int wave = threadIdx.x >> 5, lane = threadIdx.x & 31;
  int row = blockIdx.x * 8 + wave;                     // 0..2047
  int b = row >> 8, n = row & 255;
  float hreg[16];
  #pragma unroll
  for (int i = 0; i < 16; ++i) hreg[i] = hbuf[(long)row * Hh + lane + 32 * i];
  float out18[NCOL];
  #pragma unroll
  for (int o = 0; o < NCOL; ++o) {
    float acc = 0.0f;
    #pragma unroll
    for (int i = 0; i < 16; ++i) {
      int k = lane + 32 * i;
      float w = (o < 11) ? Wc[k * 11 + o] : (o < 17) ? Wb[k * 6 + (o - 11)] : Wz[k];
      acc += hreg[i] * w;
    }
    #pragma unroll
    for (int off = 16; off > 0; off >>= 1) acc += __shfl_xor(acc, off, 32);
    out18[o] = acc;
  }
  if (lane == 0) {
    #pragma unroll
    for (int o = 0; o < 11; ++o) out18[o] += bc[o];
    #pragma unroll
    for (int o = 0; o < 6; ++o) out18[11 + o] += bb[o];
    out18[17] += bz[0];
    float* ho = out + OFF_HEAD + (long)row * NCOL;
    #pragma unroll
    for (int o = 0; o < NCOL; ++o) ho[o] = out18[o];
    // softmax over 11 logits
    float mx = out18[0];
    #pragma unroll
    for (int o = 1; o < 11; ++o) mx = fmaxf(mx, out18[o]);
    float ex[11], ssum = 0.0f;
    #pragma unroll
    for (int o = 0; o < 11; ++o) { ex[o] = __expf(out18[o] - mx); ssum += ex[o]; }
    float inv = 1.0f / ssum;
    // decode box (class-agnostic)
    float bmin[3], bmax[3];
    #pragma unroll
    for (int d = 0; d < 3; ++d) {
      float pl = lbdrfu[(long)row * 6 + d]     + out18[11 + d];
      float pu = lbdrfu[(long)row * 6 + 3 + d] + out18[14 + d];
      float og = origins[(long)row * 3 + d];
      bmin[d] = og - pl;
      bmax[d] = og + pu;
    }
    for (int c = 0; c < NCLS; ++c) {
      int mcand = n * NCLS + c;
      long base = (long)b * NMS_M + mcand;
      cscores[base] = ex[c] * inv;
      float* cb = cboxes + base * 6;
      float* ob = out + OFF_BOXES + base * 6;
      #pragma unroll
      for (int d = 0; d < 3; ++d) {
        cb[d] = bmin[d]; cb[3 + d] = bmax[d];
        ob[d] = 0.5f * (bmin[d] + bmax[d]);
        ob[3 + d] = bmax[d] - bmin[d];
      }
    }
  }
}

// ---------------------------------------------------------------------------
// 5) Per-batch NMS: bitonic sort (desc) in LDS + greedy sequential suppress
// ---------------------------------------------------------------------------
__global__ void nms_kernel(const float* __restrict__ cboxes,
                           const float* __restrict__ cscores,
                           float* __restrict__ out) {
  __shared__ float ssc[NMS_P];
  __shared__ int   sid[NMS_P];
  __shared__ unsigned char ssup[NMS_M];
  __shared__ float sbox[6];
  __shared__ int   ski;
  int b = blockIdx.x, tid = threadIdx.x, nt = blockDim.x;
  const float* bs = cboxes + (long)b * NMS_M * 6;
  const float* sc = cscores + (long)b * NMS_M;
  for (int i = tid; i < NMS_P; i += nt) {
    if (i < NMS_M) { ssc[i] = sc[i]; sid[i] = i; }
    else           { ssc[i] = -3.0e38f; sid[i] = -1; }
  }
  for (int i = tid; i < NMS_M; i += nt) ssup[i] = 0;
  __syncthreads();
  // bitonic sort, descending by score
  for (int k = 2; k <= NMS_P; k <<= 1) {
    for (int j = k >> 1; j > 0; j >>= 1) {
      for (int i = tid; i < NMS_P; i += nt) {
        int ixj = i ^ j;
        if (ixj > i) {
          bool up = ((i & k) == 0);
          float a = ssc[i], c = ssc[ixj];
          bool sw = up ? (a < c) : (a > c);
          if (sw) {
            ssc[i] = c; ssc[ixj] = a;
            int t = sid[i]; sid[i] = sid[ixj]; sid[ixj] = t;
          }
        }
      }
      __syncthreads();
    }
  }
  // greedy NMS (matches reference fori_loop semantics)
  for (int i = 0; i < NMS_M; ++i) {
    __syncthreads();
    if (tid == 0) {
      int cand = sid[i];
      float si = ssc[i];
      int ki = (si > SCORE_T) && (ssup[i] == 0);
      ski = ki;
      out[OFF_SCORES  + (long)b * NMS_M + cand] = ki ? si : 0.0f;
      out[OFF_CLASSES + (long)b * NMS_M + cand] = ki ? (float)(cand % NCLS) : -1.0f;
      out[OFF_KEEP    + (long)b * NMS_M + cand] = ki ? 1.0f : 0.0f;
      if (ki) {
        #pragma unroll
        for (int d = 0; d < 6; ++d) sbox[d] = bs[(long)cand * 6 + d];
      }
    }
    __syncthreads();
    if (ski) {
      float v0 = 1.0f;
      #pragma unroll
      for (int d = 0; d < 3; ++d) v0 *= fmaxf(sbox[3 + d] - sbox[d], 0.0f);
      for (int j = i + 1 + tid; j < NMS_M; j += nt) {
        if (ssup[j]) continue;
        const float* bj = bs + (long)sid[j] * 6;
        float inter = 1.0f, v1 = 1.0f;
        #pragma unroll
        for (int d = 0; d < 3; ++d) {
          float lo = fmaxf(sbox[d],     bj[d]);
          float hi = fminf(sbox[3 + d], bj[3 + d]);
          inter *= fmaxf(hi - lo, 0.0f);
          v1    *= fmaxf(bj[3 + d] - bj[d], 0.0f);
        }
        float uni = v0 + v1 - inter;
        float iou = inter / fmaxf(uni, 1e-9f);
        if (iou > NMS_T) ssup[j] = 1;
      }
    }
  }
}

// ---------------------------------------------------------------------------
extern "C" void kernel_launch(void* const* d_in, const int* in_sizes, int n_in,
                              void* d_out, int out_size, void* d_ws, size_t ws_size,
                              hipStream_t stream) {
  const float* seed_xyz   = (const float*)d_in[0];
  const float* seed_feat  = (const float*)d_in[1];
  const float* voted      = (const float*)d_in[2];
  const int*   prop_inds  = (const int*)  d_in[3];
  const float* lbdrfu     = (const float*)d_in[4];
  const float* origins    = (const float*)d_in[5];
  const float* W1         = (const float*)d_in[6];
  const float* b1         = (const float*)d_in[7];
  const float* Wc         = (const float*)d_in[8];
  const float* bc         = (const float*)d_in[9];
  const float* Wb         = (const float*)d_in[10];
  const float* bb         = (const float*)d_in[11];
  const float* Wz         = (const float*)d_in[12];
  const float* bz         = (const float*)d_in[13];

  _Float16* featsA = (_Float16*)d_ws;                    // [2048*384] f16
  _Float16* w1t    = featsA + (long)Bb * Np * KIN;       // [512*384]  f16
  float* hbuf      = (float*)(w1t + (long)Hh * KIN);     // [2048*512] f32
  float* cboxes    = hbuf + (long)Bb * Np * Hh;          // [8*2560*6] f32
  float* cscores   = cboxes + (long)Bb * NMS_M * 6;      // [8*2560]   f32
  float* outf      = (float*)d_out;

  prep_w1_kernel<<<(Hh * KIN + 255) / 256, 256, 0, stream>>>(W1, w1t);
  gather_pool_kernel<<<Bb * Np, 256, 0, stream>>>(seed_xyz, seed_feat, voted,
                                                  prop_inds, lbdrfu, origins, featsA);
  dim3 g1((Bb * Np) / 16, Hh / 128);
  gemm1_kernel<<<g1, 256, 0, stream>>>(featsA, w1t, b1, hbuf);
  heads_kernel<<<(Bb * Np) / 8, 256, 0, stream>>>(hbuf, Wc, bc, Wb, bb, Wz, bz,
                                                  lbdrfu, origins, outf, cboxes, cscores);
  nms_kernel<<<Bb, 1024, 0, stream>>>(cboxes, cscores, outf);
}
